// BottleneckBlock3D_40991167873322
// MI455X (gfx1250) — compile-verified, matrix operand order checked by reading, not run on hardware
//
#include <hip/hip_runtime.h>
#include <hip/hip_bf16.h>

typedef __attribute__((ext_vector_type(16))) _Float16 v16h;
typedef __attribute__((ext_vector_type(8)))  float    v8f;

// ---------------------------------------------------------------- WMMA helpers

__device__ inline v8f wmma_f16(v16h a, v16h b, v8f c) {
  return __builtin_amdgcn_wmma_f32_16x16x32_f16(
      /*neg_a=*/false, a, /*neg_b=*/false, b,
      /*c_mod=*/(short)0, c, /*reuse_a=*/false, /*reuse_b=*/false);
}

// A-fragment (16x32, row-major tile[m][k]) or B-fragment from a [n][k] tile.
__device__ inline v16h load_frag16(const _Float16* p, int stride) {
  const int l  = threadIdx.x & 31;
  const int kb = (l < 16) ? 0 : 8;
  const _Float16* row = p + (size_t)(l & 15) * stride + kb;
  v16h f;
#pragma unroll
  for (int j = 0; j < 4; ++j) { f[2*j]   = row[2*j];      f[2*j+1] = row[2*j+1]; }
#pragma unroll
  for (int j = 0; j < 4; ++j) { f[8+2*j] = row[16+2*j];   f[9+2*j] = row[16+2*j+1]; }
  return f;
}

// B-fragment from a k-major tile[k][n] (stride = row pitch in halves).
__device__ inline v16h load_frag16_kmaj(const _Float16* p, int stride) {
  const int l  = threadIdx.x & 31;
  const int n  = l & 15;
  const int kb = (l < 16) ? 0 : 8;
  v16h f;
#pragma unroll
  for (int j = 0; j < 4; ++j) {
    f[2*j]   = p[(size_t)(kb + 2*j)     * stride + n];
    f[2*j+1] = p[(size_t)(kb + 2*j + 1) * stride + n];
  }
#pragma unroll
  for (int j = 0; j < 4; ++j) {
    f[8+2*j] = p[(size_t)(kb + 16 + 2*j)     * stride + n];
    f[9+2*j] = p[(size_t)(kb + 16 + 2*j + 1) * stride + n];
  }
  return f;
}

// -------------------------------------------- CDNA5 async memory->LDS helpers

// low 32 bits of a generic pointer to __shared__ == LDS byte offset
__device__ inline unsigned lds_off(const void* p) {
  return (unsigned)(unsigned long long)p;
}
// GLOBAL_LOAD_ASYNC_TO_LDS_B128: 16B memory -> LDS, tracked by ASYNCcnt
__device__ inline void async_ld16(unsigned lds, const void* g) {
  asm volatile("global_load_async_to_lds_b128 %0, %1, off"
               :: "v"(lds), "v"(g) : "memory");
}
__device__ inline void wait_async0() {
  asm volatile("s_wait_asynccnt 0" ::: "memory");
}

// ---------------------------------------------------------------- small utils

__global__ void zero_f32_kernel(float* p, int n) {
  int i = blockIdx.x * 256 + threadIdx.x;
  if (i < n) p[i] = 0.f;
}

__global__ void f32_to_f16_kernel(const float* __restrict__ s, _Float16* __restrict__ d, int n) {
  int i = blockIdx.x * 256 + threadIdx.x;
  if (i < n) d[i] = (_Float16)s[i];
}

// Expand rpb_table (2197,8) into dense bias (8,352,352); pad cols/rows = -1e30.
__global__ void bias_build_kernel(const float* __restrict__ rpb, float* __restrict__ biasF) {
  size_t idx = (size_t)blockIdx.x * 256 + threadIdx.x;
  if (idx >= (size_t)8 * 352 * 352) return;
  int m = (int)(idx % 352);
  size_t r = idx / 352;
  int n = (int)(r % 352);
  int h = (int)(r / 352);
  float v = -1e30f;
  if (n < 343 && m < 343) {
    int d1 = n / 49, h1 = (n / 7) % 7, w1 = n % 7;
    int d2 = m / 49, h2 = (m / 7) % 7, w2 = m % 7;
    int ridx = (d1 - d2 + 6) * 169 + (h1 - h2 + 6) * 13 + (w1 - w2 + 6);
    v = rpb[ridx * 8 + h];
  }
  biasF[idx] = v;
}

// ------------------------------------------------- LayerNorm (+ window partition)

// mode 0: write to window-partitioned (64,352,192) f16; mode 1: linear (t,192) f16
__global__ void __launch_bounds__(256)
ln_kernel(const float* __restrict__ x, const float* __restrict__ w,
          const float* __restrict__ b, _Float16* __restrict__ out, int mode) {
  const int wave = threadIdx.x >> 5, lane = threadIdx.x & 31;
  const int tok = blockIdx.x * 8 + wave;
  if (tok >= 21952) return;
  const float* row = x + (size_t)tok * 192;
  float v[6], s = 0.f, sq = 0.f;
#pragma unroll
  for (int i = 0; i < 6; ++i) { v[i] = row[lane + 32 * i]; s += v[i]; sq += v[i] * v[i]; }
#pragma unroll
  for (int off = 1; off < 32; off <<= 1) { s += __shfl_xor(s, off, 32); sq += __shfl_xor(sq, off, 32); }
  float mean = s * (1.f / 192.f);
  float var  = sq * (1.f / 192.f) - mean * mean;
  float inv  = rsqrtf(var + 1e-5f);
  size_t obase;
  if (mode == 0) {
    int t = tok;
    int wq = t % 28; t /= 28;
    int hq = t % 28; t /= 28;
    int dq = t % 14; int bb = t / 14;
    int win = ((bb * 2 + dq / 7) * 4 + hq / 7) * 4 + wq / 7;
    int ri  = (dq % 7) * 49 + (hq % 7) * 7 + (wq % 7);
    obase = ((size_t)win * 352 + ri) * 192;
  } else {
    obase = (size_t)tok * 192;
  }
#pragma unroll
  for (int i = 0; i < 6; ++i) {
    int c = lane + 32 * i;
    out[obase + c] = (_Float16)((v[i] - mean) * inv * w[c] + b[c]);
  }
}

// ---------------------------------------------------------------- generic WMMA GEMM
// C(M,N) = A(M,K) @ W(N,K)^T [+ bias]; cols < alphaCols scaled by alpha.

template <bool OUT_F16>
__global__ void __launch_bounds__(256)
gemm_wmma_kernel(const _Float16* __restrict__ A, const _Float16* __restrict__ W,
                 void* __restrict__ Cout, const float* __restrict__ bias,
                 int M, int N, int K, float alpha, int alphaCols) {
  __shared__ _Float16 sA[128][40];
  __shared__ _Float16 sW[64][40];
  const int tid  = threadIdx.x;
  const int wave = tid >> 5;
  const int wm   = wave >> 1, wn = wave & 1;
  const int m0   = blockIdx.y * 128;
  const int n0   = blockIdx.x * 64;
  const bool interior = (m0 + 128 <= M) && (n0 + 64 <= N);

  v8f z = {};
  v8f acc[2][2];
  acc[0][0] = z; acc[0][1] = z; acc[1][0] = z; acc[1][1] = z;

  for (int k0 = 0; k0 < K; k0 += 32) {
    if (interior) {
      // async DMA memory->LDS: A tile 128x32 halves = 512 b128 chunks,
      // W tile 64x32 halves = 256 b128 chunks; 3 per thread.
#pragma unroll
      for (int q = 0; q < 2; ++q) {
        int chunk = tid + q * 256;
        int r = chunk >> 2, part = chunk & 3;
        async_ld16(lds_off(&sA[r][part * 8]),
                   A + (size_t)(m0 + r) * K + k0 + part * 8);
      }
      {
        int r = tid >> 2, part = tid & 3;
        async_ld16(lds_off(&sW[r][part * 8]),
                   W + (size_t)(n0 + r) * K + k0 + part * 8);
      }
      wait_async0();
    } else {
      for (int i = tid; i < 128 * 32; i += 256) {
        int r = i >> 5, c = i & 31;
        int gr = m0 + r;
        sA[r][c] = (gr < M) ? A[(size_t)gr * K + k0 + c] : (_Float16)0.f;
      }
      for (int i = tid; i < 64 * 32; i += 256) {
        int r = i >> 5, c = i & 31;
        int gn = n0 + r;
        sW[r][c] = (gn < N) ? W[(size_t)gn * K + k0 + c] : (_Float16)0.f;
      }
    }
    if (k0 + 32 < K) {
      int gr = m0 + (tid >> 1);
      if (gr < M) __builtin_prefetch(&A[(size_t)gr * K + k0 + 32], 0, 1);
    }
    __syncthreads();
    v16h a0 = load_frag16(&sA[wm * 32][0], 40);
    v16h a1 = load_frag16(&sA[wm * 32 + 16][0], 40);
    v16h b0 = load_frag16(&sW[wn * 32][0], 40);
    v16h b1 = load_frag16(&sW[wn * 32 + 16][0], 40);
    acc[0][0] = wmma_f16(a0, b0, acc[0][0]);
    acc[0][1] = wmma_f16(a0, b1, acc[0][1]);
    acc[1][0] = wmma_f16(a1, b0, acc[1][0]);
    acc[1][1] = wmma_f16(a1, b1, acc[1][1]);
    __syncthreads();
  }

  const int lane = tid & 31;
  const int ln   = lane & 15;
  const int rsel = (lane < 16) ? 0 : 8;
#pragma unroll
  for (int i = 0; i < 2; ++i)
#pragma unroll
    for (int j = 0; j < 2; ++j)
#pragma unroll
      for (int e = 0; e < 8; ++e) {
        int row = m0 + wm * 32 + i * 16 + rsel + e;
        int col = n0 + wn * 32 + j * 16 + ln;
        if (row < M && col < N) {
          float v = acc[i][j][e];
          if (bias) v += bias[col];
          if (col < alphaCols) v *= alpha;
          if constexpr (OUT_F16) ((_Float16*)Cout)[(size_t)row * N + col] = (_Float16)v;
          else                   ((float*)Cout)[(size_t)row * N + col] = v;
        }
      }
}

// ---------------------------------------------------------------- flash attention
// qkv: (64,352,576) f16 (q pre-scaled), biasF: (8,352,352) f32, aout: (64,352,192) f16

__global__ void __launch_bounds__(256)
attn_kernel(const _Float16* __restrict__ qkv, const float* __restrict__ biasF,
            _Float16* __restrict__ aout) {
  __shared__ _Float16 kS[352][32];
  __shared__ _Float16 vS[352][32];
  __shared__ _Float16 pS[8][16][32];
  const int wh   = blockIdx.x;
  const int win  = wh >> 3, head = wh & 7;
  const int tid  = threadIdx.x;
  const int wave = tid >> 5, lane = tid & 31;
  const _Float16* base = qkv + (size_t)win * 352 * 576;

  // zero the pad region (d>=24 or m>=343) with ds stores
  for (int i = tid; i < 352 * 32; i += 256) {
    int m = i >> 5, d = i & 31;
    if (m >= 343 || d >= 24) { kS[m][d] = (_Float16)0.f; vS[m][d] = (_Float16)0.f; }
  }
  // async DMA the valid 343x24-half region: 3 x 16B chunks per row, per array
  for (int c = tid; c < 343 * 3; c += 256) {
    int m = c / 3, part = c % 3;
    const _Float16* gk = base + (size_t)m * 576 + 192 + head * 24 + part * 8;
    async_ld16(lds_off(&kS[m][part * 8]), gk);
    async_ld16(lds_off(&vS[m][part * 8]), gk + 192);
  }
  wait_async0();
  __syncthreads();

  const int strip = blockIdx.y * 8 + wave;
  if (strip >= 22) return;

  const int ln   = lane & 15;
  const int kb   = (lane < 16) ? 0 : 8;
  const int rsel = (lane < 16) ? 0 : 8;
  const int qr   = strip * 16 + ln;

  v16h qa;
#pragma unroll
  for (int j = 0; j < 8; ++j) {
    int kk = (j < 4) ? (kb + 2 * j) : (kb + 16 + 2 * (j - 4));
    _Float16 h0 = (_Float16)0.f, h1 = (_Float16)0.f;
    if (qr < 343) {
      const _Float16* qrow = base + (size_t)qr * 576 + head * 24;
      if (kk < 24)     h0 = qrow[kk];
      if (kk + 1 < 24) h1 = qrow[kk + 1];
    }
    qa[2 * j] = h0; qa[2 * j + 1] = h1;
  }

  float mrow[8], lrow[8];
#pragma unroll
  for (int e = 0; e < 8; ++e) { mrow[e] = -1e30f; lrow[e] = 0.f; }
  v8f o0 = {}, o1 = {};
  const float* bbase = biasF + (size_t)head * 352 * 352 + (size_t)(strip * 16) * 352;

  for (int jc = 0; jc < 11; ++jc) {
    const int mc = jc * 32;
    v16h bk0 = load_frag16(&kS[mc][0], 32);
    v16h bk1 = load_frag16(&kS[mc + 16][0], 32);
    v8f zz = {};
    v8f s0 = wmma_f16(qa, bk0, zz);
    v8f s1 = wmma_f16(qa, bk1, zz);
#pragma unroll
    for (int e = 0; e < 8; ++e) {
      const float* br = bbase + (size_t)(rsel + e) * 352 + mc + ln;
      float a0 = s0[e] + br[0];
      float a1 = s1[e] + br[16];
      float mx = fmaxf(a0, a1);
#pragma unroll
      for (int off = 1; off < 16; off <<= 1) mx = fmaxf(mx, __shfl_xor(mx, off, 32));
      float mnew = fmaxf(mrow[e], mx);
      float corr = __expf(mrow[e] - mnew);
      float p0 = __expf(a0 - mnew);
      float p1 = __expf(a1 - mnew);
      float rs = p0 + p1;
#pragma unroll
      for (int off = 1; off < 16; off <<= 1) rs += __shfl_xor(rs, off, 32);
      lrow[e] = lrow[e] * corr + rs;
      mrow[e] = mnew;
      o0[e] *= corr;
      o1[e] *= corr;
      pS[wave][rsel + e][ln]      = (_Float16)p0;
      pS[wave][rsel + e][16 + ln] = (_Float16)p1;
    }
    asm volatile("s_wait_dscnt 0" ::: "memory");  // P staging visible within wave
    v16h pa  = load_frag16(&pS[wave][0][0], 32);
    v16h bv0 = load_frag16_kmaj(&vS[mc][0], 32);
    v16h bv1 = load_frag16_kmaj(&vS[mc][16], 32);
    o0 = wmma_f16(pa, bv0, o0);
    o1 = wmma_f16(pa, bv1, o1);
  }

#pragma unroll
  for (int e = 0; e < 8; ++e) {
    int r = strip * 16 + rsel + e;
    if (r < 343) {
      float inv = 1.f / lrow[e];
      size_t ob = ((size_t)win * 352 + r) * 192 + head * 24;
      aout[ob + ln] = (_Float16)(o0[e] * inv);
      if (16 + ln < 24) aout[ob + 16 + ln] = (_Float16)(o1[e] * inv);
    }
  }
}

// ---------------------------------------------------- window-reverse + residual

__global__ void wrev_residual_kernel(const float* __restrict__ x, const float* __restrict__ p1,
                                     float* __restrict__ x1) {
  size_t idx = (size_t)blockIdx.x * 256 + threadIdx.x;
  if (idx >= (size_t)64 * 343 * 192) return;
  int c = (int)(idx % 192);
  size_t t = idx / 192;
  int row = (int)(t % 343);
  int win = (int)(t / 343);
  int wi = win & 3, hi = (win >> 2) & 3, di = (win >> 4) & 1, bb = win >> 5;
  int w0 = row % 7, h0 = (row / 7) % 7, d0 = row / 49;
  size_t tt = (((size_t)bb * 14 + di * 7 + d0) * 28 + hi * 7 + h0) * 28 + wi * 7 + w0;
  size_t o = tt * 192 + c;
  x1[o] = x[o] + p1[((size_t)win * 352 + row) * 192 + c];
}

// ---------------------------------------------------------------- group-norm bits

template <typename T>
__global__ void gn_stats_kernel(const T* __restrict__ v, float* __restrict__ st,
                                int C, int half, int S, size_t total) {
  __shared__ float ls[8];
  if (threadIdx.x < 8) ls[threadIdx.x] = 0.f;
  __syncthreads();
  size_t idx = (size_t)blockIdx.x * 256 + threadIdx.x;
  if (idx < total) {
    int c = (int)(idx % C);
    size_t t = idx / C;
    int b = (int)(t / (size_t)S);
    int slot = b * 2 + (c >= half ? 1 : 0);
    float val = (float)v[idx];
    atomicAdd(&ls[slot * 2], val);
    atomicAdd(&ls[slot * 2 + 1], val * val);
  }
  __syncthreads();
  if (threadIdx.x < 8) atomicAdd(&st[threadIdx.x], ls[threadIdx.x]);
}

__global__ void gn_finalize_kernel(const float* __restrict__ st, float* __restrict__ out, float cnt) {
  int i = threadIdx.x;
  if (i < 4) {
    float m   = st[i * 2] / cnt;
    float var = st[i * 2 + 1] / cnt - m * m;
    out[i * 2]     = m;
    out[i * 2 + 1] = rsqrtf(var + 1e-5f);
  }
}

// depthwise 3x3x3 conv; GN1 affine + relu applied to each input tap on the fly
__global__ void dwconv_kernel(const _Float16* __restrict__ e, const float* __restrict__ st,
                              const float* __restrict__ gw, const float* __restrict__ gb,
                              const float* __restrict__ dww, _Float16* __restrict__ out) {
  size_t idx = (size_t)blockIdx.x * 256 + threadIdx.x;
  if (idx >= (size_t)21952 * 768) return;
  int c = (int)(idx % 768);
  size_t t = idx / 768;
  int b  = (int)(t / 10976);
  int sp = (int)(t % 10976);
  int d  = sp / 784;
  int hh = (sp / 28) % 28;
  int ww = sp % 28;
  int slot = b * 2 + c / 384;
  float mean = st[slot * 2], inv = st[slot * 2 + 1];
  float sc  = inv * gw[c];
  float off = gb[c] - mean * sc;
  float acc = 0.f;
  for (int dz = -1; dz <= 1; ++dz) {
    int zd = d + dz; if (zd < 0 || zd >= 14) continue;
    for (int dy = -1; dy <= 1; ++dy) {
      int yh = hh + dy; if (yh < 0 || yh >= 28) continue;
      for (int dx = -1; dx <= 1; ++dx) {
        int xw2 = ww + dx; if (xw2 < 0 || xw2 >= 28) continue;
        size_t tn = (size_t)b * 10976 + (size_t)(zd * 784 + yh * 28 + xw2);
        float v = (float)e[tn * 768 + c];
        v = fmaxf(v * sc + off, 0.f);
        acc += dww[c * 27 + (dz + 1) * 9 + (dy + 1) * 3 + (dx + 1)] * v;
      }
    }
  }
  out[idx] = (_Float16)acc;
}

// GN2 + relu, store g; deterministic per-(b,c) SE squeeze means
__global__ void __launch_bounds__(256)
gn2_se_kernel(const _Float16* __restrict__ dv, const float* __restrict__ st,
              const float* __restrict__ gw, const float* __restrict__ gb,
              _Float16* __restrict__ g, float* __restrict__ sesum) {
  __shared__ float red[256];
  int bc = blockIdx.x;
  int b = bc / 768, c = bc % 768;
  int slot = b * 2 + c / 384;
  float mean = st[slot * 2], inv = st[slot * 2 + 1];
  float sc  = inv * gw[c];
  float off = gb[c] - mean * sc;
  float sum = 0.f;
  for (int i = threadIdx.x; i < 10976; i += 256) {
    size_t t = (size_t)b * 10976 + i;
    float v = (float)dv[t * 768 + c];
    v = fmaxf(v * sc + off, 0.f);
    g[t * 768 + c] = (_Float16)v;
    sum += v;
  }
  red[threadIdx.x] = sum;
  __syncthreads();
  for (int s = 128; s > 0; s >>= 1) {
    if (threadIdx.x < s) red[threadIdx.x] += red[threadIdx.x + s];
    __syncthreads();
  }
  if (threadIdx.x == 0) sesum[bc] = red[0] / 10976.f;
}

__global__ void se_kernel(const float* __restrict__ sesum,
                          const float* __restrict__ w1, const float* __restrict__ b1,
                          const float* __restrict__ w2, const float* __restrict__ b2,
                          float* __restrict__ s_sig) {
  __shared__ float sh1[192];
  for (int b = 0; b < 2; ++b) {
    for (int o = threadIdx.x; o < 192; o += 256) {
      float a = b1[o];
      for (int e = 0; e < 768; ++e) a += sesum[b * 768 + e] * w1[o * 768 + e];
      sh1[o] = fmaxf(a, 0.f);
    }
    __syncthreads();
    for (int e = threadIdx.x; e < 768; e += 256) {
      float a = b2[e];
      for (int o = 0; o < 192; ++o) a += sh1[o] * w2[e * 192 + o];
      s_sig[b * 768 + e] = 1.f / (1.f + __expf(-a));
    }
    __syncthreads();
  }
}

__global__ void scale_g_kernel(_Float16* __restrict__ g, const float* __restrict__ s_sig) {
  size_t idx = (size_t)blockIdx.x * 256 + threadIdx.x;
  if (idx >= (size_t)21952 * 768) return;
  int c = (int)(idx % 768);
  size_t t = idx / 768;
  int b = (int)(t / 10976);
  g[idx] = (_Float16)((float)g[idx] * s_sig[b * 768 + c]);
}

__global__ void final_kernel(const float* __restrict__ x1, const float* __restrict__ p,
                             const float* __restrict__ st, const float* __restrict__ gw,
                             const float* __restrict__ gb, float* __restrict__ out) {
  size_t idx = (size_t)blockIdx.x * 256 + threadIdx.x;
  if (idx >= (size_t)21952 * 192) return;
  int c = (int)(idx % 192);
  size_t t = idx / 192;
  int b = (int)(t / 10976);
  int slot = b * 2 + c / 96;
  float mean = st[slot * 2], inv = st[slot * 2 + 1];
  out[idx] = x1[idx] + (p[idx] - mean) * inv * gw[c] + gb[c];
}

// ---------------------------------------------------------------- launcher

extern "C" void kernel_launch(void* const* d_in, const int* in_sizes, int n_in,
                              void* d_out, int out_size, void* d_ws, size_t ws_size,
                              hipStream_t stream) {
  (void)in_sizes; (void)n_in; (void)out_size;
  const float* x        = (const float*)d_in[0];
  const float* norm1_w  = (const float*)d_in[2];
  const float* norm1_b  = (const float*)d_in[3];
  const float* qkv_w    = (const float*)d_in[4];
  const float* qkv_b    = (const float*)d_in[5];
  const float* rpb      = (const float*)d_in[6];
  const float* proj_w   = (const float*)d_in[7];
  const float* proj_b   = (const float*)d_in[8];
  const float* norm2_w  = (const float*)d_in[9];
  const float* norm2_b  = (const float*)d_in[10];
  const float* expand_w = (const float*)d_in[11];
  const float* gn1_w    = (const float*)d_in[12];
  const float* gn1_b    = (const float*)d_in[13];
  const float* dw_w     = (const float*)d_in[14];
  const float* gn2_w    = (const float*)d_in[15];
  const float* gn2_b    = (const float*)d_in[16];
  const float* se_w1    = (const float*)d_in[17];
  const float* se_b1    = (const float*)d_in[18];
  const float* se_w2    = (const float*)d_in[19];
  const float* se_b2    = (const float*)d_in[20];
  const float* pw_w     = (const float*)d_in[21];
  const float* gn3_w    = (const float*)d_in[22];
  const float* gn3_b    = (const float*)d_in[23];

  // workspace layout (bytes)
  constexpr size_t OFF_WQKV  = 0;
  constexpr size_t SZ_WQKV   = (size_t)576 * 192 * 2;
  constexpr size_t OFF_WPROJ = OFF_WQKV + SZ_WQKV;
  constexpr size_t SZ_WPROJ  = (size_t)192 * 192 * 2;
  constexpr size_t OFF_WEXP  = OFF_WPROJ + SZ_WPROJ;
  constexpr size_t SZ_WEXP   = (size_t)768 * 192 * 2;
  constexpr size_t OFF_WPW   = OFF_WEXP + SZ_WEXP;
  constexpr size_t SZ_WPW    = (size_t)192 * 768 * 2;
  constexpr size_t OFF_BIAS  = OFF_WPW + SZ_WPW;
  constexpr size_t SZ_BIAS   = (size_t)8 * 352 * 352 * 4;
  constexpr size_t OFF_X1    = OFF_BIAS + SZ_BIAS;
  constexpr size_t SZ_X1     = (size_t)21952 * 192 * 4;
  constexpr size_t OFF_P1    = OFF_X1 + SZ_X1;
  constexpr size_t SZ_P1     = (size_t)22528 * 192 * 4;
  constexpr size_t OFF_STATS = OFF_P1 + SZ_P1;
  constexpr size_t SZ_STATS  = 65536;
  constexpr size_t OFF_POOL  = OFF_STATS + SZ_STATS;
  constexpr size_t OFF_XW    = OFF_POOL;                 // attention phase
  constexpr size_t SZ_XW     = (size_t)64 * 352 * 192 * 2;
  constexpr size_t OFF_QKV   = OFF_XW + SZ_XW;
  constexpr size_t SZ_QKV    = (size_t)64 * 352 * 576 * 2;
  constexpr size_t OFF_AOUT  = OFF_QKV + SZ_QKV;
  constexpr size_t SZ_AOUT   = SZ_XW;
  constexpr size_t OFF_XN2   = OFF_POOL;                 // mbconv phase (reuses pool)
  constexpr size_t SZ_XN2    = (size_t)21952 * 192 * 2;
  constexpr size_t OFF_E     = OFF_XN2 + SZ_XN2;
  constexpr size_t SZ_E      = (size_t)21952 * 768 * 2;
  constexpr size_t OFF_DV    = OFF_AOUT + SZ_AOUT;
  constexpr size_t SZ_DV     = SZ_E;
  constexpr size_t WS_NEED   = OFF_DV + SZ_DV;
  if (ws_size < WS_NEED) return;

  char* ws = (char*)d_ws;
  _Float16* wqkv16  = (_Float16*)(ws + OFF_WQKV);
  _Float16* wproj16 = (_Float16*)(ws + OFF_WPROJ);
  _Float16* wexp16  = (_Float16*)(ws + OFF_WEXP);
  _Float16* wpw16   = (_Float16*)(ws + OFF_WPW);
  float*    biasF   = (float*)(ws + OFF_BIAS);
  float*    x1      = (float*)(ws + OFF_X1);
  float*    p1      = (float*)(ws + OFF_P1);
  float*    st_raw  = (float*)(ws + OFF_STATS);
  float*    st_fin  = (float*)(ws + OFF_STATS + 256);
  float*    sesum   = (float*)(ws + OFF_STATS + 1024);
  float*    s_sig   = (float*)(ws + OFF_STATS + 8192);
  _Float16* xw16    = (_Float16*)(ws + OFF_XW);
  _Float16* qkv16   = (_Float16*)(ws + OFF_QKV);
  _Float16* aout16  = (_Float16*)(ws + OFF_AOUT);
  _Float16* xn216   = (_Float16*)(ws + OFF_XN2);
  _Float16* e16     = (_Float16*)(ws + OFF_E);
  _Float16* dv16    = (_Float16*)(ws + OFF_DV);
  float*    out     = (float*)d_out;

  const float qscale = 0.20412414523193154f;  // 24^-0.5

  // prep
  zero_f32_kernel<<<1, 64, 0, stream>>>(st_raw, 64);
  f32_to_f16_kernel<<<(576 * 192 + 255) / 256, 256, 0, stream>>>(qkv_w, wqkv16, 576 * 192);
  f32_to_f16_kernel<<<(192 * 192 + 255) / 256, 256, 0, stream>>>(proj_w, wproj16, 192 * 192);
  f32_to_f16_kernel<<<(768 * 192 + 255) / 256, 256, 0, stream>>>(expand_w, wexp16, 768 * 192);
  f32_to_f16_kernel<<<(192 * 768 + 255) / 256, 256, 0, stream>>>(pw_w, wpw16, 192 * 768);
  bias_build_kernel<<<(8 * 352 * 352 + 255) / 256, 256, 0, stream>>>(rpb, biasF);

  // part 1: attention
  ln_kernel<<<2744, 256, 0, stream>>>(x, norm1_w, norm1_b, xw16, 0);
  gemm_wmma_kernel<true><<<dim3(9, 176), 256, 0, stream>>>(
      xw16, wqkv16, qkv16, qkv_b, 22528, 576, 192, qscale, 192);
  attn_kernel<<<dim3(512, 3), 256, 0, stream>>>(qkv16, biasF, aout16);
  gemm_wmma_kernel<false><<<dim3(3, 176), 256, 0, stream>>>(
      aout16, wproj16, p1, proj_b, 22528, 192, 192, 1.f, 0);
  wrev_residual_kernel<<<16464, 256, 0, stream>>>(x, p1, x1);

  // part 2: MBConv
  ln_kernel<<<2744, 256, 0, stream>>>(x1, norm2_w, norm2_b, xn216, 1);
  gemm_wmma_kernel<true><<<dim3(12, 172), 256, 0, stream>>>(
      xn216, wexp16, e16, nullptr, 21952, 768, 192, 1.f, 0);
  gn_stats_kernel<_Float16><<<65856, 256, 0, stream>>>(
      e16, st_raw + 0, 768, 384, 10976, (size_t)21952 * 768);
  gn_finalize_kernel<<<1, 32, 0, stream>>>(st_raw + 0, st_fin + 0, 10976.f * 384.f);
  dwconv_kernel<<<65856, 256, 0, stream>>>(e16, st_fin + 0, gn1_w, gn1_b, dw_w, dv16);
  gn_stats_kernel<_Float16><<<65856, 256, 0, stream>>>(
      dv16, st_raw + 8, 768, 384, 10976, (size_t)21952 * 768);
  gn_finalize_kernel<<<1, 32, 0, stream>>>(st_raw + 8, st_fin + 8, 10976.f * 384.f);
  gn2_se_kernel<<<1536, 256, 0, stream>>>(dv16, st_fin + 8, gn2_w, gn2_b, e16, sesum);
  se_kernel<<<1, 256, 0, stream>>>(sesum, se_w1, se_b1, se_w2, se_b2, s_sig);
  scale_g_kernel<<<65856, 256, 0, stream>>>(e16, s_sig);
  gemm_wmma_kernel<false><<<dim3(3, 172), 256, 0, stream>>>(
      e16, wpw16, p1, nullptr, 21952, 192, 768, 1.f, 0);
  gn_stats_kernel<float><<<16464, 256, 0, stream>>>(
      p1, st_raw + 16, 192, 96, 10976, (size_t)21952 * 192);
  gn_finalize_kernel<<<1, 32, 0, stream>>>(st_raw + 16, st_fin + 16, 10976.f * 96.f);
  final_kernel<<<16464, 256, 0, stream>>>(x1, p1, st_fin + 16, gn3_w, gn3_b, out);
}
